// CausalSelfAttention_3135326126336
// MI455X (gfx1250) — compile-verified, matrix operand order checked by reading, not run on hardware
//
#include <hip/hip_runtime.h>
#include <hip/hip_bf16.h>

typedef unsigned short u16;
typedef unsigned int   u32;
typedef __attribute__((ext_vector_type(16))) __bf16 v16bf;
typedef __attribute__((ext_vector_type(8)))  float  v8f;
typedef int v4i_ __attribute__((vector_size(16)));

#define B_  4
#define T_  2048
#define C_  1024
#define NH_ 16
#define HD_ 64
#define M_  (B_ * T_)   // 8192

#if defined(__has_builtin)
#if __has_builtin(__builtin_amdgcn_global_load_async_to_lds_b128) && \
    __has_builtin(__builtin_amdgcn_s_wait_asynccnt)
#define HAVE_ASYNC_LDS 1
#endif
#endif

__device__ __forceinline__ u16 f2bf(float f) {
    u32 u = __float_as_uint(f);
    u32 r = (u + 0x7FFFu + ((u >> 16) & 1u)) >> 16;   // round-to-nearest-even
    return (u16)r;
}

union Frag16 { uint4 q[2]; v16bf v; };   // 16 bf16 = one WMMA A/B fragment per lane

// ---------------------------------------------------------------- convert ----
__global__ void k_f32_to_bf16(const float* __restrict__ in, u16* __restrict__ out, int n4) {
    int i = blockIdx.x * blockDim.x + threadIdx.x;
    if (i >= n4) return;
    float4 f = ((const float4*)in)[i];
    u32 lo = (u32)f2bf(f.x) | ((u32)f2bf(f.y) << 16);
    u32 hi = (u32)f2bf(f.z) | ((u32)f2bf(f.w) << 16);
    ((uint2*)out)[i] = make_uint2(lo, hi);
}

// ------------------------------------------------- W [K][N] f32 -> Wt [N][K] bf16
__global__ void k_transpose_w(const float* __restrict__ W, u16* __restrict__ Wt) {
    __shared__ float tile[32][33];
    int bx = blockIdx.x * 32, by = blockIdx.y * 32;
    int tx = threadIdx.x & 31, ty = threadIdx.x >> 5;      // 256 threads: ty 0..7
#pragma unroll
    for (int i = 0; i < 4; ++i) {
        int r = ty + i * 8;
        tile[r][tx] = W[(size_t)(by + r) * C_ + bx + tx];
    }
    __syncthreads();
#pragma unroll
    for (int i = 0; i < 4; ++i) {
        int r = ty + i * 8;                                // n = bx + r, k = by + tx
        Wt[(size_t)(bx + r) * C_ + by + tx] = f2bf(tile[tx][r]);
    }
}

// ---------------------------------------------------------------- GEMM ----
// C[M,N] = A[M,K] * Bt[N,K]^T, M=8192, N=K=1024.
// Block tile 128x256, 256 threads = 8 waves (2x4), wave tile 64x64 (acc[4][4]).
// Double-buffered LDS; async global->LDS copies when the toolchain exposes them.
// MODE 0: write bf16 [B,H,T,D]; MODE 1: write bf16 [B,H,D,T]; MODE 2: write f32 [M,N].
#define GBM 128
#define GBN 256
#define GBK 32

template <int MODE>
__global__ __launch_bounds__(256) void k_gemm(const u16* __restrict__ A,
                                              const u16* __restrict__ Bt,
                                              void* __restrict__ outp) {
    __shared__ __attribute__((aligned(16))) u16 sA[2][GBM * GBK];   // 2 x 8 KB
    __shared__ __attribute__((aligned(16))) u16 sB[2][GBN * GBK];   // 2 x 16 KB
    const int tid  = threadIdx.x;
    const int wid  = tid >> 5, lane = tid & 31;
    const int l15  = lane & 15, half = lane >> 4;
    const int m0   = blockIdx.y * GBM, n0 = blockIdx.x * GBN;
    const int wm   = (wid >> 2) * 64,  wn = (wid & 3) * 64;
    const int NK   = C_ / GBK;   // 32

    const v8f z8 = {0.f, 0.f, 0.f, 0.f, 0.f, 0.f, 0.f, 0.f};
    v8f acc[4][4];
#pragma unroll
    for (int fm = 0; fm < 4; ++fm)
#pragma unroll
        for (int fn = 0; fn < 4; ++fn) acc[fm][fn] = z8;

#if HAVE_ASYNC_LDS
#define ASYNC_CP16(g, l)                                                        \
    __builtin_amdgcn_global_load_async_to_lds_b128(                             \
        (__attribute__((address_space(1))) v4i_*)(g),                           \
        (__attribute__((address_space(3))) v4i_*)(l), 0, 0)
    // prologue: stage k-step 0 into buffer 0
    {
        const int k0 = 0;
#pragma unroll
        for (int c = 0; c < 2; ++c) {
            int chunk = tid + 256 * c, row = chunk >> 2, col = (chunk & 3) * 8;
            ASYNC_CP16(A + (size_t)(m0 + row) * C_ + k0 + col, &sA[0][row * GBK + col]);
        }
#pragma unroll
        for (int c = 0; c < 4; ++c) {
            int chunk = tid + 256 * c, row = chunk >> 2, col = (chunk & 3) * 8;
            ASYNC_CP16(Bt + (size_t)(n0 + row) * C_ + k0 + col, &sB[0][row * GBK + col]);
        }
        __builtin_amdgcn_s_wait_asynccnt(0);
    }
    __syncthreads();

    for (int kt = 0; kt < NK; ++kt) {
        const int cur = kt & 1;
        if (kt + 1 < NK) {                       // async-stage next tile into other buffer
            const int k0 = (kt + 1) * GBK;
#pragma unroll
            for (int c = 0; c < 2; ++c) {
                int chunk = tid + 256 * c, row = chunk >> 2, col = (chunk & 3) * 8;
                ASYNC_CP16(A + (size_t)(m0 + row) * C_ + k0 + col, &sA[cur ^ 1][row * GBK + col]);
            }
#pragma unroll
            for (int c = 0; c < 4; ++c) {
                int chunk = tid + 256 * c, row = chunk >> 2, col = (chunk & 3) * 8;
                ASYNC_CP16(Bt + (size_t)(n0 + row) * C_ + k0 + col, &sB[cur ^ 1][row * GBK + col]);
            }
        }
        const u16* cA = sA[cur];
        const u16* cB = sB[cur];
        Frag16 af[4];
#pragma unroll
        for (int fm = 0; fm < 4; ++fm) {
            const u16* p = cA + (wm + fm * 16 + l15) * GBK + half * 8;
            af[fm].q[0] = *(const uint4*)p;
            af[fm].q[1] = *(const uint4*)(p + 16);
        }
#pragma unroll
        for (int fn = 0; fn < 4; ++fn) {
            Frag16 bfr;
            const u16* p = cB + (wn + fn * 16 + l15) * GBK + half * 8;
            bfr.q[0] = *(const uint4*)p;
            bfr.q[1] = *(const uint4*)(p + 16);
#pragma unroll
            for (int fm = 0; fm < 4; ++fm)
                acc[fm][fn] = __builtin_amdgcn_wmma_f32_16x16x32_bf16(
                    false, af[fm].v, false, bfr.v, (short)0, acc[fm][fn], false, false);
        }
        __builtin_amdgcn_s_wait_asynccnt(0);
        __syncthreads();
    }
#undef ASYNC_CP16
#else
    // fallback: register-prefetch double buffering
    uint4 ra[2], rb[4];
    {
        const int k0 = 0;
#pragma unroll
        for (int c = 0; c < 2; ++c) {
            int chunk = tid + 256 * c, row = chunk >> 2, col = (chunk & 3) * 8;
            ra[c] = *(const uint4*)(A + (size_t)(m0 + row) * C_ + k0 + col);
        }
#pragma unroll
        for (int c = 0; c < 4; ++c) {
            int chunk = tid + 256 * c, row = chunk >> 2, col = (chunk & 3) * 8;
            rb[c] = *(const uint4*)(Bt + (size_t)(n0 + row) * C_ + k0 + col);
        }
    }
    for (int kt = 0; kt < NK; ++kt) {
        const int cur = kt & 1;
#pragma unroll
        for (int c = 0; c < 2; ++c) {
            int chunk = tid + 256 * c, row = chunk >> 2, col = (chunk & 3) * 8;
            *(uint4*)(&sA[cur][row * GBK + col]) = ra[c];
        }
#pragma unroll
        for (int c = 0; c < 4; ++c) {
            int chunk = tid + 256 * c, row = chunk >> 2, col = (chunk & 3) * 8;
            *(uint4*)(&sB[cur][row * GBK + col]) = rb[c];
        }
        __syncthreads();
        if (kt + 1 < NK) {
            const int k0 = (kt + 1) * GBK;
#pragma unroll
            for (int c = 0; c < 2; ++c) {
                int chunk = tid + 256 * c, row = chunk >> 2, col = (chunk & 3) * 8;
                ra[c] = *(const uint4*)(A + (size_t)(m0 + row) * C_ + k0 + col);
            }
#pragma unroll
            for (int c = 0; c < 4; ++c) {
                int chunk = tid + 256 * c, row = chunk >> 2, col = (chunk & 3) * 8;
                rb[c] = *(const uint4*)(Bt + (size_t)(n0 + row) * C_ + k0 + col);
            }
        }
        const u16* cA = sA[cur];
        const u16* cB = sB[cur];
        Frag16 af[4];
#pragma unroll
        for (int fm = 0; fm < 4; ++fm) {
            const u16* p = cA + (wm + fm * 16 + l15) * GBK + half * 8;
            af[fm].q[0] = *(const uint4*)p;
            af[fm].q[1] = *(const uint4*)(p + 16);
        }
#pragma unroll
        for (int fn = 0; fn < 4; ++fn) {
            Frag16 bfr;
            const u16* p = cB + (wn + fn * 16 + l15) * GBK + half * 8;
            bfr.q[0] = *(const uint4*)p;
            bfr.q[1] = *(const uint4*)(p + 16);
#pragma unroll
            for (int fm = 0; fm < 4; ++fm)
                acc[fm][fn] = __builtin_amdgcn_wmma_f32_16x16x32_bf16(
                    false, af[fm].v, false, bfr.v, (short)0, acc[fm][fn], false, false);
        }
        __syncthreads();
    }
#endif

#pragma unroll
    for (int fm = 0; fm < 4; ++fm)
#pragma unroll
        for (int fn = 0; fn < 4; ++fn)
#pragma unroll
            for (int j = 0; j < 8; ++j) {
                int gm = m0 + wm + fm * 16 + j + half * 8;   // row: b*T + t
                int gn = n0 + wn + fn * 16 + l15;            // col: h*HD + d
                float v = acc[fm][fn][j];
                if (MODE == 2) {
                    ((float*)outp)[(size_t)gm * C_ + gn] = v;
                } else {
                    int b = gm >> 11, t = gm & (T_ - 1);
                    int h = gn >> 6,  d = gn & (HD_ - 1);
                    size_t idx = (MODE == 0)
                        ? ((size_t)(b * NH_ + h) * T_ + t) * HD_ + d
                        : ((size_t)(b * NH_ + h) * HD_ + d) * T_ + t;
                    ((u16*)outp)[idx] = f2bf(v);
                }
            }
}

// ---------------------------------------------------------- flash attention ----
// grid (B*NH, T/64), 128 threads = 4 waves; each wave owns 16 query rows.
__global__ __launch_bounds__(128) void k_attn(const u16* __restrict__ Q,
                                              const u16* __restrict__ K,
                                              const u16* __restrict__ Vt,
                                              u16* __restrict__ Y) {
    __shared__ __attribute__((aligned(16))) u16 sP[4][16 * 32];
    const int tid  = threadIdx.x;
    const int wid  = tid >> 5, lane = tid & 31;
    const int l15  = lane & 15, half = lane >> 4;
    const int bh   = blockIdx.x;
    const int t0   = blockIdx.y * 64 + wid * 16;
    const size_t baseQK = (size_t)bh * T_ * HD_;
    const size_t baseV  = (size_t)bh * HD_ * T_;
    u16* myP = sP[wid];

    Frag16 qf[2];
    {
        const u16* qrow = Q + baseQK + (size_t)(t0 + l15) * HD_;
#pragma unroll
        for (int ks = 0; ks < 2; ++ks) {
            const u16* p = qrow + ks * 32 + half * 8;
            qf[ks].q[0] = *(const uint4*)p;
            qf[ks].q[1] = *(const uint4*)(p + 16);
        }
    }

    const v8f z8 = {0.f, 0.f, 0.f, 0.f, 0.f, 0.f, 0.f, 0.f};
    v8f acc[4];
#pragma unroll
    for (int f = 0; f < 4; ++f) acc[f] = z8;
    float rmax[8], rsum[8];
#pragma unroll
    for (int j = 0; j < 8; ++j) { rmax[j] = -3.0e38f; rsum[j] = 0.f; }

    const int nblk = (t0 + 16 + 31) >> 5;
    for (int sb = 0; sb < nblk; ++sb) {
        const int s0 = sb * 32;
        v8f sfr[2] = {z8, z8};
#pragma unroll
        for (int f = 0; f < 2; ++f) {
            const u16* krow = K + baseQK + (size_t)(s0 + f * 16 + l15) * HD_;
#pragma unroll
            for (int ks = 0; ks < 2; ++ks) {
                Frag16 kf;
                const u16* p = krow + ks * 32 + half * 8;
                kf.q[0] = *(const uint4*)p;
                kf.q[1] = *(const uint4*)(p + 16);
                sfr[f] = __builtin_amdgcn_wmma_f32_16x16x32_bf16(
                    false, qf[ks].v, false, kf.v, (short)0, sfr[f], false, false);
            }
        }
        const bool onDiag = (s0 + 31 > t0);
#pragma unroll
        for (int j = 0; j < 8; ++j) {
            int t = t0 + j + half * 8;
            float x0 = sfr[0][j] * 0.125f;          // 1/sqrt(64)
            float x1 = sfr[1][j] * 0.125f;
            if (onDiag) {
                if (s0 + l15 > t)      x0 = -3.0e38f;
                if (s0 + 16 + l15 > t) x1 = -3.0e38f;
            }
            float bm = fmaxf(x0, x1);
#pragma unroll
            for (int off = 1; off < 16; off <<= 1) bm = fmaxf(bm, __shfl_xor(bm, off, 32));
            float nm   = fmaxf(rmax[j], bm);
            float corr = exp2f((rmax[j] - nm) * 1.44269504f);
            float p0   = exp2f((x0 - nm) * 1.44269504f);
            float p1   = exp2f((x1 - nm) * 1.44269504f);
            float ps   = p0 + p1;
#pragma unroll
            for (int off = 1; off < 16; off <<= 1) ps += __shfl_xor(ps, off, 32);
            rsum[j] = rsum[j] * corr + ps;
            rmax[j] = nm;
#pragma unroll
            for (int f = 0; f < 4; ++f) acc[f][j] *= corr;
            int m = j + half * 8;
            myP[m * 32 + l15]      = f2bf(p0);      // C-layout -> row-major P in LDS
            myP[m * 32 + 16 + l15] = f2bf(p1);
        }
        Frag16 pf;                                   // reload P as A-fragment
        const u16* pp = myP + l15 * 32 + half * 8;
        pf.q[0] = *(const uint4*)pp;
        pf.q[1] = *(const uint4*)(pp + 16);
#pragma unroll
        for (int f = 0; f < 4; ++f) {
            Frag16 vf;
            const u16* vrow = Vt + baseV + (size_t)(f * 16 + l15) * T_ + s0 + half * 8;
            vf.q[0] = *(const uint4*)vrow;
            vf.q[1] = *(const uint4*)(vrow + 16);
            acc[f] = __builtin_amdgcn_wmma_f32_16x16x32_bf16(
                false, pf.v, false, vf.v, (short)0, acc[f], false, false);
        }
    }

    const int b = bh >> 4, h = bh & 15;
#pragma unroll
    for (int f = 0; f < 4; ++f)
#pragma unroll
        for (int j = 0; j < 8; ++j) {
            int t = t0 + j + half * 8;
            int d = f * 16 + l15;
            float y = acc[f][j] / rsum[j];
            Y[((size_t)b * T_ + t) * C_ + h * HD_ + d] = f2bf(y);   // [B,T,H,D] == [M,C]
        }
}

// ---------------------------------------------------------------- launch ----
extern "C" void kernel_launch(void* const* d_in, const int* in_sizes, int n_in,
                              void* d_out, int out_size, void* d_ws, size_t ws_size,
                              hipStream_t stream) {
    const float* x  = (const float*)d_in[0];
    const float* Wq = (const float*)d_in[1];
    const float* Wk = (const float*)d_in[2];
    const float* Wv = (const float*)d_in[3];
    const float* Wp = (const float*)d_in[4];
    float* out = (float*)d_out;

    char* ws = (char*)d_ws;
    size_t o = 0;
    u16* Xbf = (u16*)(ws + o); o += (size_t)M_ * C_ * sizeof(u16);   // 16 MB
    u16* Wqt = (u16*)(ws + o); o += (size_t)C_ * C_ * sizeof(u16);   // 2 MB
    u16* Wkt = (u16*)(ws + o); o += (size_t)C_ * C_ * sizeof(u16);
    u16* Wvt = (u16*)(ws + o); o += (size_t)C_ * C_ * sizeof(u16);
    u16* Wpt = (u16*)(ws + o); o += (size_t)C_ * C_ * sizeof(u16);
    u16* Qb  = (u16*)(ws + o); o += (size_t)M_ * C_ * sizeof(u16);   // [B,H,T,D]
    u16* Kb  = (u16*)(ws + o); o += (size_t)M_ * C_ * sizeof(u16);   // [B,H,T,D]
    u16* Vtb = (u16*)(ws + o); o += (size_t)M_ * C_ * sizeof(u16);   // [B,H,D,T]
    u16* Ybf = (u16*)(ws + o); o += (size_t)M_ * C_ * sizeof(u16);   // [B,T,H,D]

    int n4 = (M_ * C_) / 4;
    k_f32_to_bf16<<<(n4 + 255) / 256, 256, 0, stream>>>(x, Xbf, n4);

    dim3 tg(C_ / 32, C_ / 32);
    k_transpose_w<<<tg, 256, 0, stream>>>(Wq, Wqt);
    k_transpose_w<<<tg, 256, 0, stream>>>(Wk, Wkt);
    k_transpose_w<<<tg, 256, 0, stream>>>(Wv, Wvt);
    k_transpose_w<<<tg, 256, 0, stream>>>(Wp, Wpt);

    dim3 gg(C_ / GBN, M_ / GBM);                                     // (4, 64)
    k_gemm<0><<<gg, 256, 0, stream>>>(Xbf, Wqt, (void*)Qb);
    k_gemm<0><<<gg, 256, 0, stream>>>(Xbf, Wkt, (void*)Kb);
    k_gemm<1><<<gg, 256, 0, stream>>>(Xbf, Wvt, (void*)Vtb);

    k_attn<<<dim3(B_ * NH_, T_ / 64), 128, 0, stream>>>(Qb, Kb, Vtb, Ybf);

    k_gemm<2><<<gg, 256, 0, stream>>>(Ybf, Wpt, (void*)out);
}
